// GraphormerEncoder_54863912239505
// MI455X (gfx1250) — matrix operand order repaired and checked by reading, not compiled, not run on hardware
//
#include <hip/hip_runtime.h>
#include <hip/hip_bf16.h>

// ---------------- problem constants ----------------
constexpr int kN  = 1024;   // nodes
constexpr int kE  = 16384;  // edges
constexpr int kD  = 256;    // embed dim
constexpr int kH  = 8;      // heads
constexpr int kHD = 32;     // head dim
constexpr int kFF = 1024;   // ffn dim
constexpr int kL  = 4;      // layers
constexpr int FWT = 32;     // Floyd-Warshall tile

typedef __attribute__((ext_vector_type(16))) _Float16 v16h;
typedef __attribute__((ext_vector_type(8)))  _Float16 v8h;
typedef __attribute__((ext_vector_type(8)))  float    v8f;

static __device__ __forceinline__ v16h cat8(v8h lo, v8h hi) {
  return __builtin_shufflevector(lo, hi, 0, 1, 2, 3, 4, 5, 6, 7,
                                 8, 9, 10, 11, 12, 13, 14, 15);
}

// ---------------- small utility kernels ----------------

// wt[n*K + k] = (f16) W[k*Nn + n]; rows n in [Nn, NnPad) zero-padded.
// Makes every GEMM B-operand column-major => per-lane contiguous fragments.
__global__ void transpose_f32_to_f16_kernel(const float* __restrict__ W,
                                            _Float16* __restrict__ wt,
                                            int K, int Nn, int NnPad) {
  int i = blockIdx.x * 256 + threadIdx.x;
  if (i >= NnPad * K) return;
  int n = i / K, k = i - n * K;
  wt[i] = (n < Nn) ? (_Float16)W[(size_t)k * Nn + n] : (_Float16)0.0f;
}

// x[n,d] = node_features @ W_node + b_node + node_encoding
__global__ void node_proj_kernel(const float* __restrict__ nf,
                                 const float* __restrict__ W,
                                 const float* __restrict__ bias,
                                 const float* __restrict__ enc,
                                 float* __restrict__ x) {
  int row = blockIdx.x, d = threadIdx.x;
  float s = bias[d] + enc[(size_t)row * kD + d];
#pragma unroll
  for (int k = 0; k < 12; ++k) s += nf[row * 12 + k] * W[k * kD + d];
  x[(size_t)row * kD + d] = s;
}

// e[i,d] = edge_attr @ W_edge + b_edge  (stored directly as f16 for WMMA A)
__global__ void edge_proj_kernel(const float* __restrict__ ea,
                                 const float* __restrict__ W,
                                 const float* __restrict__ bias,
                                 _Float16* __restrict__ ebf) {
  int row = blockIdx.x, d = threadIdx.x;
  float s = bias[d];
#pragma unroll
  for (int k = 0; k < 12; ++k) s += ea[row * 12 + k] * W[k * kD + d];
  ebf[(size_t)row * kD + d] = (_Float16)s;
}

// ---------------- Floyd-Warshall ----------------

__global__ void dist_init_kernel(float* __restrict__ dist) {
  int idx = blockIdx.x * 256 + threadIdx.x;
  int i = idx >> 10, j = idx & (kN - 1);
  dist[idx] = (i == j) ? 0.0f : __builtin_inff();
}

__global__ void dist_edges_kernel(const int* __restrict__ ei,
                                  float* __restrict__ dist) {
  int e = blockIdx.x * 256 + threadIdx.x;
  if (e >= kE) return;
  int s = ei[e], t = ei[kE + e];
  if (s != t) {
    dist[(size_t)s * kN + t] = 1.0f;
    dist[(size_t)t * kN + s] = 1.0f;
  }
}

__global__ void fw_phase1(float* __restrict__ dist, int r) {
  __shared__ float tile[FWT][FWT + 1];
  int i = threadIdx.y, j = threadIdx.x, b = r * FWT;
  tile[i][j] = dist[(size_t)(b + i) * kN + b + j];
  __syncthreads();
  for (int k = 0; k < FWT; ++k) {
    float nv = tile[i][k] + tile[k][j];
    if (nv < tile[i][j]) tile[i][j] = nv;
    __syncthreads();
  }
  dist[(size_t)(b + i) * kN + b + j] = tile[i][j];
}

__global__ void fw_phase2(float* __restrict__ dist, int r) {
  int c = blockIdx.x;
  if (c == r) return;
  __shared__ float piv[FWT][FWT + 1], cur[FWT][FWT + 1];
  int i = threadIdx.y, j = threadIdx.x, b = r * FWT, cb = c * FWT;
  piv[i][j] = dist[(size_t)(b + i) * kN + b + j];
  if (blockIdx.y == 0) {  // tile (r, c)
    cur[i][j] = dist[(size_t)(b + i) * kN + cb + j];
    __syncthreads();
    for (int k = 0; k < FWT; ++k) {
      float nv = piv[i][k] + cur[k][j];
      if (nv < cur[i][j]) cur[i][j] = nv;
      __syncthreads();
    }
    dist[(size_t)(b + i) * kN + cb + j] = cur[i][j];
  } else {                // tile (c, r)
    cur[i][j] = dist[(size_t)(cb + i) * kN + b + j];
    __syncthreads();
    for (int k = 0; k < FWT; ++k) {
      float nv = cur[i][k] + piv[k][j];
      if (nv < cur[i][j]) cur[i][j] = nv;
      __syncthreads();
    }
    dist[(size_t)(cb + i) * kN + b + j] = cur[i][j];
  }
}

__global__ void fw_phase3(float* __restrict__ dist, int r) {
  int bi = blockIdx.y, bj = blockIdx.x;
  if (bi == r || bj == r) return;
  __shared__ float rowT[FWT][FWT + 1], colT[FWT][FWT + 1];
  int i = threadIdx.y, j = threadIdx.x, b = r * FWT;
  rowT[i][j] = dist[(size_t)(bi * FWT + i) * kN + b + j];
  colT[i][j] = dist[(size_t)(b + i) * kN + bj * FWT + j];
  float cur = dist[(size_t)(bi * FWT + i) * kN + bj * FWT + j];
  __syncthreads();
#pragma unroll 8
  for (int k = 0; k < FWT; ++k) cur = fminf(cur, rowT[i][k] + colT[k][j]);
  dist[(size_t)(bi * FWT + i) * kN + bj * FWT + j] = cur;
}

__global__ void didx_kernel(const float* __restrict__ dist,
                            int* __restrict__ didx) {
  int idx = blockIdx.x * 256 + threadIdx.x;
  didx[idx] = (int)fminf(dist[idx], 99.0f);  // inf -> N -> clipped to 99
}

// ---------------- layernorm / softmax ----------------

__global__ void layernorm_kernel(const float* __restrict__ x,
                                 const float* __restrict__ w,
                                 const float* __restrict__ b,
                                 _Float16* __restrict__ outh,
                                 float* __restrict__ outf) {
  __shared__ float red[kD];
  int t = threadIdx.x, row = blockIdx.x;
  float v = x[(size_t)row * kD + t];
  red[t] = v; __syncthreads();
  for (int s = kD / 2; s > 0; s >>= 1) {
    if (t < s) red[t] += red[t + s];
    __syncthreads();
  }
  float mean = red[0] * (1.0f / kD); __syncthreads();
  float c = v - mean;
  red[t] = c * c; __syncthreads();
  for (int s = kD / 2; s > 0; s >>= 1) {
    if (t < s) red[t] += red[t + s];
    __syncthreads();
  }
  float var = red[0] * (1.0f / kD);
  float y = c * rsqrtf(var + 1e-5f) * w[t] + b[t];
  if (outh) outh[(size_t)row * kD + t] = (_Float16)y;
  if (outf) outf[(size_t)row * kD + t] = y;
}

// softmax over rows of length kN; writes f16 probs for the next WMMA
__global__ void softmax_kernel(const float* __restrict__ S,
                               _Float16* __restrict__ P) {
  __shared__ float red[256];
  int t = threadIdx.x;
  size_t base = (size_t)blockIdx.x * kN;
  float vals[4], mx = -1e30f;
#pragma unroll
  for (int i = 0; i < 4; ++i) {
    vals[i] = S[base + t + i * 256];
    mx = fmaxf(mx, vals[i]);
  }
  red[t] = mx; __syncthreads();
  for (int s = 128; s > 0; s >>= 1) {
    if (t < s) red[t] = fmaxf(red[t], red[t + s]);
    __syncthreads();
  }
  mx = red[0]; __syncthreads();
  float sum = 0.0f;
#pragma unroll
  for (int i = 0; i < 4; ++i) { vals[i] = __expf(vals[i] - mx); sum += vals[i]; }
  red[t] = sum; __syncthreads();
  for (int s = 128; s > 0; s >>= 1) {
    if (t < s) red[t] += red[t + s];
    __syncthreads();
  }
  float inv = 1.0f / red[0];
#pragma unroll
  for (int i = 0; i < 4; ++i)
    P[base + t + i * 256] = (_Float16)(vals[i] * inv);
}

// scatter per-edge attention bias: S[h][src][tgt] += eenc[e][h]
__global__ void edge_bias_scatter(const int* __restrict__ ei,
                                  const float* __restrict__ eenc,
                                  float* __restrict__ S) {
  int idx = blockIdx.x * 256 + threadIdx.x;
  if (idx >= kE * kH) return;
  int e = idx >> 3, h = idx & 7;
  int s = ei[e], t = ei[kE + e];
  atomicAdd(&S[((size_t)h * kN + s) * kN + t], eenc[(size_t)e * kH + h]);
}

// ---------------- generic batched WMMA GEMM (register-blocked) ----------------
// C[z] = act(alpha * A[z] @ B[z] + bias) (+resid);  f16 in, f32 accum.
// A row-major (lda), B COLUMN-major (ldb = stride between columns) so every
// lane's 16-element fragment is two contiguous 128-bit loads.
// Block = 128 threads (4 waves). Each wave owns a 16 x (16*NB) output strip:
// one A fragment is reused against NB B fragments -> NB back-to-back WMMAs
// per K-step, 4x less A traffic for NB=4.
// cht: write Ch transposed (Ch[col*ldc + row]) -- used to produce V^T.
template <int NB>
__global__ void gemm_wmma_kernel(const _Float16* __restrict__ A, int lda, long long batchA,
                                 const _Float16* __restrict__ B, int ldb, long long batchB,
                                 float* __restrict__ Cf, _Float16* __restrict__ Ch,
                                 int ldc, long long batchC, int cht,
                                 const float* __restrict__ bias,
                                 const float* __restrict__ resid,
                                 int M, int Nn, int K, float alpha, int act) {
  int lane = threadIdx.x & 31;
  int wv   = threadIdx.x >> 5;
  int z    = blockIdx.z;
  int tm = (blockIdx.y * 4 + wv) * 16;
  int tn = blockIdx.x * (16 * NB);
  int l15 = lane & 15;
  int kg  = lane >> 4;  // lane half selects K sub-range
  const _Float16* ap = A + (long long)z * batchA + (size_t)(tm + l15) * lda + kg * 8;
  const _Float16* bp = B + (long long)z * batchB + (size_t)(tn + l15) * ldb + kg * 8;
  v8f acc[NB];
#pragma unroll
  for (int j = 0; j < NB; ++j) acc[j] = (v8f){};
#pragma unroll 2
  for (int k0 = 0; k0 < K; k0 += 32) {
    v8h a0 = *(const v8h*)(ap);       // K = k0+kg*8 .. +7
    v8h a1 = *(const v8h*)(ap + 16);  // K = k0+16+kg*8 .. +7
    v16h a = cat8(a0, a1);
#pragma unroll
    for (int j = 0; j < NB; ++j) {
      const _Float16* bj = bp + (size_t)(16 * j) * ldb;
      v8h b0 = *(const v8h*)(bj);      // K = k0+kg*8 .. +7
      v8h b1 = *(const v8h*)(bj + 16); // K = k0+16+kg*8 .. +7
      acc[j] = __builtin_amdgcn_wmma_f32_16x16x32_f16(false, a, false, cat8(b0, b1),
                                                      (short)0, acc[j], false, false);
    }
    ap += 32;
    bp += 32;
  }
  int mb = tm + kg * 8;
  long long cz = (long long)z * batchC;
#pragma unroll
  for (int j = 0; j < NB; ++j) {
    int bcol = tn + 16 * j + l15;
    if (bcol >= Nn) continue;  // only ragged case is Nn=8 (edge encoder, NB=1)
#pragma unroll
    for (int r = 0; r < 8; ++r) {
      float val = acc[j][r] * alpha;
      if (bias) val += bias[bcol];
      if (act == 1) val = 0.5f * val * (1.0f + erff(val * 0.70710678118654752f));
      if (cht) {
        Ch[(size_t)cz + (size_t)bcol * ldc + (mb + r)] = (_Float16)val;
      } else {
        size_t ci = (size_t)cz + (size_t)(mb + r) * ldc + bcol;
        if (resid) val += resid[ci];
        if (Cf) Cf[ci] = val;
        if (Ch) Ch[ci] = (_Float16)val;
      }
    }
  }
}

// ---------------- attention scores: S = scale*Q.K^T + dist_bias ----------------
// One wave per 16x16 tile; K = HD = 32 -> exactly one WMMA per tile.
// K-operand fragments are contiguous per lane (row-major keys act column-major).
__global__ void attn_scores_kernel(const _Float16* __restrict__ q,
                                   const _Float16* __restrict__ k,
                                   const int* __restrict__ didx,
                                   const float* __restrict__ dtab, // [100,H]
                                   float* __restrict__ S, float scale) {
  int lane = threadIdx.x & 31;
  int wv   = threadIdx.x >> 5;
  int h    = blockIdx.z;
  int tN = (blockIdx.y * 8 + wv) * 16;  // n tile (query rows)
  int tM = blockIdx.x * 16;             // m tile (key cols)
  int l15 = lane & 15;
  int kg  = lane >> 4;
  const _Float16* ap = q + (size_t)(tN + l15) * kD + h * kHD + kg * 8;
  const _Float16* bp = k + (size_t)(tM + l15) * kD + h * kHD + kg * 8;
  v8h a0 = *(const v8h*)(ap);
  v8h a1 = *(const v8h*)(ap + 16);
  v8h b0 = *(const v8h*)(bp);
  v8h b1 = *(const v8h*)(bp + 16);
  v8f acc = {};
  acc = __builtin_amdgcn_wmma_f32_16x16x32_f16(false, cat8(a0, a1),
                                               false, cat8(b0, b1),
                                               (short)0, acc, false, false);
  int m  = tM + l15;
  int nb = tN + kg * 8;
#pragma unroll
  for (int r = 0; r < 8; ++r) {
    int n = nb + r;
    float biasv = dtab[didx[(size_t)n * kN + m] * kH + h];
    S[((size_t)h * kN + n) * kN + m] = acc[r] * scale + biasv;
  }
}

// ---------------- host orchestration ----------------

extern "C" void kernel_launch(void* const* d_in, const int* in_sizes, int n_in,
                              void* d_out, int out_size, void* d_ws, size_t ws_size,
                              hipStream_t stream) {
  (void)in_sizes; (void)n_in; (void)out_size; (void)ws_size;
  const float* node_features = (const float*)d_in[0];
  const int*   edge_index    = (const int*)d_in[1];
  const float* edge_attr     = (const float*)d_in[2];
  const float* W_node = (const float*)d_in[3];
  const float* b_node = (const float*)d_in[4];
  const float* W_edge = (const float*)d_in[5];
  const float* b_edge = (const float*)d_in[6];
  const float* node_encoding = (const float*)d_in[7];
  const float* Wq = (const float*)d_in[8];
  const float* bq = (const float*)d_in[9];
  const float* Wk = (const float*)d_in[10];
  const float* bk = (const float*)d_in[11];
  const float* Wv = (const float*)d_in[12];
  const float* bv = (const float*)d_in[13];
  const float* Wo = (const float*)d_in[14];
  const float* bo = (const float*)d_in[15];
  const float* dist_table = (const float*)d_in[16];
  const float* W_eenc = (const float*)d_in[17];
  const float* b_eenc = (const float*)d_in[18];
  const float* ln1_w = (const float*)d_in[19];
  const float* ln1_b = (const float*)d_in[20];
  const float* ln2_w = (const float*)d_in[21];
  const float* ln2_b = (const float*)d_in[22];
  const float* W1 = (const float*)d_in[23];
  const float* b1 = (const float*)d_in[24];
  const float* W2 = (const float*)d_in[25];
  const float* b2 = (const float*)d_in[26];
  const float* lnf_w = (const float*)d_in[27];
  const float* lnf_b = (const float*)d_in[28];

  char* ws = (char*)d_ws;
  size_t off = 0;
  auto take = [&](size_t bytes) -> void* {
    void* p = ws + off;
    off += (bytes + 255) & ~(size_t)255;
    return p;
  };
  float*    x0   = (float*)take((size_t)kN * kD * 4);
  float*    x1   = (float*)take((size_t)kN * kD * 4);
  _Float16* hbf  = (_Float16*)take((size_t)kN * kD * 2);
  _Float16* qh   = (_Float16*)take((size_t)kN * kD * 2);
  _Float16* kh   = (_Float16*)take((size_t)kN * kD * 2);
  _Float16* vt   = (_Float16*)take((size_t)kD * kN * 2);  // V transposed [dcol][n]
  _Float16* oh   = (_Float16*)take((size_t)kN * kD * 2);
  float*    S    = (float*)take((size_t)kH * kN * kN * 4);
  _Float16* P    = (_Float16*)take((size_t)kH * kN * kN * 2);
  float*    dist = (float*)take((size_t)kN * kN * 4);
  int*      didx = (int*)take((size_t)kN * kN * 4);
  _Float16* ebf  = (_Float16*)take((size_t)kE * kD * 2);
  float*    eenc = (float*)take((size_t)kE * kH * 4);
  _Float16* ffnh = (_Float16*)take((size_t)kN * kFF * 2);
  _Float16* whalf= (_Float16*)take((size_t)kD * kFF * 2);  // transposed weights

  const float scale = 0.17677669529663687f;  // HD^-0.5

  // encoders
  node_proj_kernel<<<kN, kD, 0, stream>>>(node_features, W_node, b_node,
                                          node_encoding, x0);
  edge_proj_kernel<<<kE, kD, 0, stream>>>(edge_attr, W_edge, b_edge, ebf);

  // Floyd-Warshall (blocked, 3-phase)
  dist_init_kernel<<<kN * kN / 256, 256, 0, stream>>>(dist);
  dist_edges_kernel<<<kE / 256, 256, 0, stream>>>(edge_index, dist);
  for (int r = 0; r < kN / FWT; ++r) {
    fw_phase1<<<1, dim3(FWT, FWT), 0, stream>>>(dist, r);
    fw_phase2<<<dim3(kN / FWT, 2), dim3(FWT, FWT), 0, stream>>>(dist, r);
    fw_phase3<<<dim3(kN / FWT, kN / FWT), dim3(FWT, FWT), 0, stream>>>(dist, r);
  }
  didx_kernel<<<kN * kN / 256, 256, 0, stream>>>(dist, didx);

  for (int l = 0; l < kL; ++l) {
    // LN1 -> f16 activations
    layernorm_kernel<<<kN, kD, 0, stream>>>(x0, ln1_w + l * kD, ln1_b + l * kD,
                                            hbf, nullptr);
    // Q,K,V projections (WMMA); weights transposed to column-major f16
    transpose_f32_to_f16_kernel<<<kD * kD / 256, 256, 0, stream>>>(
        Wq + (size_t)l * kD * kD, whalf, kD, kD, kD);
    gemm_wmma_kernel<4><<<dim3(kD / 64, kN / 64, 1), 128, 0, stream>>>(
        hbf, kD, 0, whalf, kD, 0, nullptr, qh, kD, 0, 0,
        bq + l * kD, nullptr, kN, kD, kD, 1.0f, 0);
    transpose_f32_to_f16_kernel<<<kD * kD / 256, 256, 0, stream>>>(
        Wk + (size_t)l * kD * kD, whalf, kD, kD, kD);
    gemm_wmma_kernel<4><<<dim3(kD / 64, kN / 64, 1), 128, 0, stream>>>(
        hbf, kD, 0, whalf, kD, 0, nullptr, kh, kD, 0, 0,
        bk + l * kD, nullptr, kN, kD, kD, 1.0f, 0);
    transpose_f32_to_f16_kernel<<<kD * kD / 256, 256, 0, stream>>>(
        Wv + (size_t)l * kD * kD, whalf, kD, kD, kD);
    gemm_wmma_kernel<4><<<dim3(kD / 64, kN / 64, 1), 128, 0, stream>>>(
        hbf, kD, 0, whalf, kD, 0, nullptr, vt, kN, 0, /*cht=*/1,
        bv + l * kD, nullptr, kN, kD, kD, 1.0f, 0);

    // scores = scale*QK^T + distance bias (fused gather)
    attn_scores_kernel<<<dim3(kN / 16, kN / 128, kH), 256, 0, stream>>>(
        qh, kh, didx, dist_table + (size_t)l * 100 * kH, S, scale);

    // per-edge bias: eenc = e @ W_eenc + b_eenc, scatter-add into S
    transpose_f32_to_f16_kernel<<<16 * kD / 256, 256, 0, stream>>>(
        W_eenc + (size_t)l * kD * kH, whalf, kD, kH, 16);  // pad 8 -> 16 cols
    gemm_wmma_kernel<1><<<dim3(1, kE / 64, 1), 128, 0, stream>>>(
        ebf, kD, 0, whalf, kD, 0, eenc, nullptr, kH, 0, 0,
        b_eenc + l * kH, nullptr, kE, kH, kD, 1.0f, 0);
    edge_bias_scatter<<<kE * kH / 256, 256, 0, stream>>>(edge_index, eenc, S);

    // softmax rows -> f16 probs
    softmax_kernel<<<kH * kN, 256, 0, stream>>>(S, P);

    // O_h = P_h @ V_h  (batched over heads; B = V^T, column-major contiguous)
    gemm_wmma_kernel<2><<<dim3(1, kN / 64, kH), 128, 0, stream>>>(
        P, kN, (long long)kN * kN, vt, kN, (long long)kHD * kN,
        nullptr, oh, kD, (long long)kHD, 0, nullptr, nullptr,
        kN, kHD, kN, 1.0f, 0);

    // out projection + residual -> x1
    transpose_f32_to_f16_kernel<<<kD * kD / 256, 256, 0, stream>>>(
        Wo + (size_t)l * kD * kD, whalf, kD, kD, kD);
    gemm_wmma_kernel<4><<<dim3(kD / 64, kN / 64, 1), 128, 0, stream>>>(
        oh, kD, 0, whalf, kD, 0, x1, nullptr, kD, 0, 0,
        bo + l * kD, x0, kN, kD, kD, 1.0f, 0);

    // LN2 -> f16
    layernorm_kernel<<<kN, kD, 0, stream>>>(x1, ln2_w + l * kD, ln2_b + l * kD,
                                            hbf, nullptr);
    // FFN1 (+GELU) -> f16
    transpose_f32_to_f16_kernel<<<kFF * kD / 256, 256, 0, stream>>>(
        W1 + (size_t)l * kD * kFF, whalf, kD, kFF, kFF);
    gemm_wmma_kernel<4><<<dim3(kFF / 64, kN / 64, 1), 128, 0, stream>>>(
        hbf, kD, 0, whalf, kD, 0, nullptr, ffnh, kFF, 0, 0,
        b1 + l * kFF, nullptr, kN, kFF, kD, 1.0f, 1);
    // FFN2 + residual -> x0 (current x lives in x0 at every layer boundary)
    transpose_f32_to_f16_kernel<<<kD * kFF / 256, 256, 0, stream>>>(
        W2 + (size_t)l * kFF * kD, whalf, kFF, kD, kD);
    gemm_wmma_kernel<4><<<dim3(kD / 64, kN / 64, 1), 128, 0, stream>>>(
        ffnh, kFF, 0, whalf, kFF, 0, x0, nullptr, kD, 0, 0,
        b2 + l * kD, x1, kN, kD, kFF, 1.0f, 0);
  }

  // final layernorm -> f32 output
  layernorm_kernel<<<kN, kD, 0, stream>>>(x0, lnf_w, lnf_b, nullptr,
                                          (float*)d_out);
}
